// MultiHeadAttention_88536455839974
// MI455X (gfx1250) — matrix operand order repaired and checked by reading, not compiled, not run on hardware
//
#include <hip/hip_runtime.h>

// MHA forward for MI455X (gfx1250): bf16 WMMA everywhere, fp32 accumulate.
// B=2, S=2048, D=1024, H=16, HD=64.

#define Bv  2
#define Sv  2048
#define Dv  1024
#define Hv  16
#define HDv 64
#define Mv  (Bv * Sv)   // 4096 rows in all GEMMs

typedef unsigned short u16;
typedef __attribute__((ext_vector_type(16))) __bf16 v16bf;
typedef __attribute__((ext_vector_type(8)))  float  v8f;
typedef __attribute__((ext_vector_type(16))) u16    v16u;
typedef __attribute__((ext_vector_type(8)))  u16    v8u;

// ---------- helpers ----------

__device__ __forceinline__ u16 f2bf(float f) {
  unsigned u = __builtin_bit_cast(unsigned, f);
  u += 0x7fffu + ((u >> 16) & 1u);   // round-to-nearest-even (inputs finite)
  return (u16)(u >> 16);
}

__device__ __forceinline__ v16u pack16(v8u lo, v8u hi) {
  v16u r;
#pragma unroll
  for (int i = 0; i < 8; ++i) { r[i] = lo[i]; r[i + 8] = hi[i]; }
  return r;
}

__device__ __forceinline__ v8f wmma_bf16(v16u a, v16u b, v8f c) {
  return __builtin_amdgcn_wmma_f32_16x16x32_bf16(
      false, __builtin_bit_cast(v16bf, a),
      false, __builtin_bit_cast(v16bf, b),
      (short)0, c, false, false);
}

// ---------- kernel 1: fp32 -> bf16(u16) cast ----------

__global__ void cast_bf16_kernel(const float* __restrict__ src,
                                 u16* __restrict__ dst, int n) {
  int i = blockIdx.x * blockDim.x + threadIdx.x;
  if (i < n) dst[i] = f2bf(src[i]);
}

// ---------- kernel 2: GEMM  C[m][n] = sum_k A[m][k]*W[n][k] + bias[n] ----------
// A: Mv x Dv bf16 row-major, W: Dv x Dv bf16 row-major (used transposed),
// out fp32 Mv x Dv. Block = 256 threads = 8 waves (4 in M x 2 in N),
// wave tile = 32x64 (2x4 WMMA, 8 wmma per 32-k step). Block tile 128x128.
// Grid = (Mv/128, Dv/128).

__global__ __launch_bounds__(256) void gemm_nt_kernel(
    const u16* __restrict__ A, const u16* __restrict__ W,
    const float* __restrict__ bias, float* __restrict__ C) {
  const int tid  = threadIdx.x;
  const int w    = tid >> 5;
  const int l    = tid & 31;
  const int half = l >> 4;
  const int lq   = l & 15;
  const int bm   = blockIdx.x * 128 + (w >> 1) * 32;
  const int bn   = blockIdx.y * 128 + (w & 1) * 64;

  v8f acc[2][4] = {};

  for (int k0 = 0; k0 < Dv; k0 += 32) {
    v16u af[2], bfr[4];
#pragma unroll
    for (int t = 0; t < 2; ++t) {
      // A fragment (16x32, rows bm + t*16): lane row = base row + lq,
      // k elements = half*16 + 0..15
      const u16* pa = A + (size_t)(bm + t * 16 + lq) * Dv + k0 + half * 16;
      af[t] = *(const v16u*)pa;
      if (t == 0) __builtin_prefetch(pa + 128, 0, 3);  // ~4 k-steps ahead
    }
#pragma unroll
    for (int j = 0; j < 4; ++j) {
      // B fragment (32x16, cols bn + j*16): lane col = base col + lq,
      // k elements = half*16 + 0..15 (contiguous 32B from weight row)
      const u16* pb = W + (size_t)(bn + j * 16 + lq) * Dv + k0 + half * 16;
      bfr[j] = *(const v16u*)pb;
      if (j == 0) __builtin_prefetch(pb + 128, 0, 3);
    }
#pragma unroll
    for (int i = 0; i < 2; ++i)
#pragma unroll
      for (int j = 0; j < 4; ++j)
        acc[i][j] = wmma_bf16(af[i], bfr[j], acc[i][j]);
  }

#pragma unroll
  for (int i = 0; i < 2; ++i)
#pragma unroll
    for (int j = 0; j < 4; ++j) {
      const int col = bn + j * 16 + lq;
      const float bvl = bias[col];
#pragma unroll
      for (int e = 0; e < 8; ++e) {
        const int row = bm + i * 16 + e + 8 * half;  // C layout
        C[(size_t)row * Dv + col] = acc[i][j][e] + bvl;
      }
    }
}

// ---------- kernel 3: RoPE + cast / V transpose ----------
// mode 0: rope(src)[B,S,H,HD] -> dst bf16 [B,H,S,HD]
// mode 2: src[B,S,H,HD]       -> dst bf16 [B,H,HD,S]  (V transposed)

__global__ void rope_cast_kernel(const float* __restrict__ src,
                                 u16* __restrict__ dst, int mode) {
  int idx = blockIdx.x * blockDim.x + threadIdx.x;  // B*S*H*(HD/2)
  if (idx >= Bv * Sv * Hv * (HDv / 2)) return;
  const int i = idx & 31;
  const int h = (idx >> 5) & (Hv - 1);
  const int s = (idx >> 9) & (Sv - 1);
  const int b = idx >> 20;

  const float* p = src + ((size_t)(b * Sv + s) * Dv) + h * HDv;
  const float x1 = p[i];
  const float x2 = p[i + 32];

  if (mode == 2) {
    u16* d = dst + ((size_t)((b * Hv + h) * HDv)) * Sv + s;
    d[(size_t)i * Sv]        = f2bf(x1);
    d[(size_t)(i + 32) * Sv] = f2bf(x2);
  } else {
    // inv_freq = 10000^(-2i/HD)
    const float inv_freq = __expf(-(float)(2 * i) * (9.2103403719761836f / 64.0f));
    const float theta = (float)s * inv_freq;
    float sn, cs;
    sincosf(theta, &sn, &cs);
    u16* d = dst + ((size_t)((b * Hv + h) * Sv + s)) * HDv;
    d[i]      = f2bf(x1 * cs - x2 * sn);
    d[i + 32] = f2bf(x1 * sn + x2 * cs);
  }
}

// ---------- kernel 4: causal flash attention ----------
// qb,kb: [B,H,S,64] bf16 ; vt: [B,H,64,S] bf16 ; ctx: [B,S,D] bf16.
// One wave per 16-query tile; 32-key tiles with online softmax.
// K tiles (contiguous 4KB in the [B,H,S,64] layout) are staged into per-wave
// LDS with global_load_async_to_lds_b128, software-pipelined one tile ahead.

__global__ __launch_bounds__(256) void attn_kernel(
    const u16* __restrict__ qb, const u16* __restrict__ kb,
    const u16* __restrict__ vt, u16* __restrict__ ctx) {
  __shared__ __align__(64) u16 pls[8][16][32];  // per-wave P tile (bf16), 8KB
  __shared__ __align__(64) u16 kls[8][32][64];  // per-wave K tile (bf16), 32KB

  const int tid  = threadIdx.x;
  const int w    = tid >> 5;
  const int l    = tid & 31;
  const int half = l >> 4;
  const int lq   = l & 15;

  const int g     = blockIdx.x * 8 + w;           // 0 .. B*H*(S/16)-1
  const int b     = g >> 11;                      // / (H * S/16)
  const int h     = (g >> 7) & (Hv - 1);
  const int qbase = (g & 127) * 16;

  const u16* Q  = qb + ((size_t)(b * Hv + h) * Sv) * HDv;
  const u16* Kp = kb + ((size_t)(b * Hv + h) * Sv) * HDv;
  const u16* Vp = vt + ((size_t)(b * Hv + h) * HDv) * Sv;

  // async-copy one contiguous 4KB K tile (keys kb0..kb0+31) into this wave's
  // LDS buffer: 8 x b128 per lane, both sides advance by the same inst offset.
  const unsigned kls_base = (unsigned)(uintptr_t)(&kls[w][0][0]) + (unsigned)(l * 16);
  auto issue_ktile = [&](int kb0) {
    const unsigned long long ga =
        (unsigned long long)(uintptr_t)(Kp + (size_t)kb0 * HDv) + (unsigned)(l * 16);
    asm volatile(
        "global_load_async_to_lds_b128 %0, %1, off\n\t"
        "global_load_async_to_lds_b128 %0, %1, off offset:512\n\t"
        "global_load_async_to_lds_b128 %0, %1, off offset:1024\n\t"
        "global_load_async_to_lds_b128 %0, %1, off offset:1536\n\t"
        "global_load_async_to_lds_b128 %0, %1, off offset:2048\n\t"
        "global_load_async_to_lds_b128 %0, %1, off offset:2560\n\t"
        "global_load_async_to_lds_b128 %0, %1, off offset:3072\n\t"
        "global_load_async_to_lds_b128 %0, %1, off offset:3584"
        :: "v"(kls_base), "v"(ga) : "memory");
  };

  // Q fragments: A-layout, K-dim = 64 -> two 16x32 fragments
  v16u aQ[2];
#pragma unroll
  for (int ks = 0; ks < 2; ++ks) {
    const u16* p = Q + (size_t)(qbase + lq) * HDv + ks * 32 + half * 16;
    aQ[ks] = *(const v16u*)p;
  }

  v8f o[4] = {};
  float m[8], ll[8];
#pragma unroll
  for (int e = 0; e < 8; ++e) { m[e] = -__builtin_huge_valf(); ll[e] = 0.0f; }

  issue_ktile(0);  // prime the pipeline

  for (int kb0 = 0; kb0 < qbase + 16; kb0 += 32) {
    // ---- pull staged K tile into B-fragments, then start next tile ----
    asm volatile("s_wait_asynccnt 0" ::: "memory");
    v16u kf[2][2];
#pragma unroll
    for (int nt = 0; nt < 2; ++nt)
#pragma unroll
      for (int ks = 0; ks < 2; ++ks)
        kf[nt][ks] = *(const v16u*)&kls[w][nt * 16 + lq][ks * 32 + half * 16];
    asm volatile("s_wait_dscnt 0" ::: "memory");   // frags in VGPRs before WAR
    if (kb0 + 32 < qbase + 16) issue_ktile(kb0 + 32);

    // ---- scores: S(16x32) = Q(16x64) @ K^T ----
    v8f sc[2] = {};
#pragma unroll
    for (int nt = 0; nt < 2; ++nt)
#pragma unroll
      for (int ks = 0; ks < 2; ++ks)
        sc[nt] = wmma_bf16(aQ[ks], kf[nt][ks], sc[nt]);

    // ---- online softmax (per row = e + 8*half, cols striped over lanes) ----
#pragma unroll
    for (int e = 0; e < 8; ++e) {
      const int qrow = qbase + e + 8 * half;
      float s0 = sc[0][e] * 0.125f;                         // 1/sqrt(64)
      float s1 = sc[1][e] * 0.125f;
      if (kb0 + lq > qrow)      s0 = -1e30f;                // causal mask
      if (kb0 + 16 + lq > qrow) s1 = -1e30f;
      float rm = fmaxf(s0, s1);
#pragma unroll
      for (int off = 1; off < 16; off <<= 1)
        rm = fmaxf(rm, __shfl_xor(rm, off, 32));
      const float mnew = fmaxf(m[e], rm);
      const float scl  = __expf(m[e] - mnew);
      const float p0   = __expf(s0 - mnew);
      const float p1   = __expf(s1 - mnew);
      float rs = p0 + p1;
#pragma unroll
      for (int off = 1; off < 16; off <<= 1)
        rs += __shfl_xor(rs, off, 32);
      ll[e] = ll[e] * scl + rs;
      m[e]  = mnew;
#pragma unroll
      for (int t = 0; t < 4; ++t) o[t][e] *= scl;
      pls[w][e + 8 * half][lq]      = f2bf(p0);
      pls[w][e + 8 * half][lq + 16] = f2bf(p1);
    }

    // same-wave LDS RAW: DS ops are in-order per wave; wait to be safe
    asm volatile("s_wait_dscnt 0" ::: "memory");

    // reload P in A-layout (row = lq, keys half*16+0..15)
    const v16u pa = *(const v16u*)&pls[w][lq][half * 16];

    // ---- O(16x64) += P(16x32) @ V(32x64), V pre-transposed ----
#pragma unroll
    for (int t = 0; t < 4; ++t) {
      const u16* p = Vp + (size_t)(t * 16 + lq) * Sv + kb0 + half * 16;
      o[t] = wmma_bf16(pa, *(const v16u*)p, o[t]);
    }
  }

  // normalize + store ctx as bf16 [B,S,H*HD]
#pragma unroll
  for (int e = 0; e < 8; ++e) {
    const float inv = 1.0f / ll[e];
    const int srow  = qbase + e + 8 * half;
#pragma unroll
    for (int t = 0; t < 4; ++t) {
      ctx[((size_t)(b * Sv + srow)) * Dv + h * HDv + t * 16 + lq] =
          f2bf(o[t][e] * inv);
    }
  }
}

// ---------- launch ----------
// Workspace layout (needs 64 MiB):
//  [0,8M)    xb   bf16 x
//  [8M,16M)  wq/wk/wv/wo bf16 (2 MiB each)
//  [16M,32M) fp32 scratch (reused for Q, then K, then V projections)
//  [32M,40M) qb ; [40M,48M) kbuf ; [48M,56M) vt ; [56M,64M) ctx

extern "C" void kernel_launch(void* const* d_in, const int* in_sizes, int n_in,
                              void* d_out, int out_size, void* d_ws, size_t ws_size,
                              hipStream_t stream) {
  const float* x  = (const float*)d_in[0];
  const float* Wq = (const float*)d_in[1];
  const float* bq = (const float*)d_in[2];
  const float* Wk = (const float*)d_in[3];
  const float* bk = (const float*)d_in[4];
  const float* Wv = (const float*)d_in[5];
  const float* bv = (const float*)d_in[6];
  const float* Wo = (const float*)d_in[7];
  const float* bo = (const float*)d_in[8];
  // d_in[9] = mask: causal mask applied analytically in-kernel.

  char* ws = (char*)d_ws;
  const size_t MB = 1024 * 1024;
  u16*   xb   = (u16*)(ws);
  u16*   wqb  = (u16*)(ws + 8 * MB);
  u16*   wkb  = (u16*)(ws + 10 * MB);
  u16*   wvb  = (u16*)(ws + 12 * MB);
  u16*   wob  = (u16*)(ws + 14 * MB);
  float* tmp  = (float*)(ws + 16 * MB);
  u16*   qbuf = (u16*)(ws + 32 * MB);
  u16*   kbuf = (u16*)(ws + 40 * MB);
  u16*   vtb  = (u16*)(ws + 48 * MB);
  u16*   ctx  = (u16*)(ws + 56 * MB);

  const int nX = Bv * Sv * Dv;   // 4194304
  const int nW = Dv * Dv;        // 1048576
  cast_bf16_kernel<<<(nX + 255) / 256, 256, 0, stream>>>(x,  xb,  nX);
  cast_bf16_kernel<<<(nW + 255) / 256, 256, 0, stream>>>(Wq, wqb, nW);
  cast_bf16_kernel<<<(nW + 255) / 256, 256, 0, stream>>>(Wk, wkb, nW);
  cast_bf16_kernel<<<(nW + 255) / 256, 256, 0, stream>>>(Wv, wvb, nW);
  cast_bf16_kernel<<<(nW + 255) / 256, 256, 0, stream>>>(Wo, wob, nW);

  const dim3 ggrid(Mv / 128, Dv / 128);     // (32, 8)
  const int nR = Bv * Sv * Hv * (HDv / 2);  // 2097152

  gemm_nt_kernel<<<ggrid, 256, 0, stream>>>(xb, wqb, bq, tmp);
  rope_cast_kernel<<<(nR + 255) / 256, 256, 0, stream>>>(tmp, qbuf, 0);

  gemm_nt_kernel<<<ggrid, 256, 0, stream>>>(xb, wkb, bk, tmp);
  rope_cast_kernel<<<(nR + 255) / 256, 256, 0, stream>>>(tmp, kbuf, 0);

  gemm_nt_kernel<<<ggrid, 256, 0, stream>>>(xb, wvb, bv, tmp);
  rope_cast_kernel<<<(nR + 255) / 256, 256, 0, stream>>>(tmp, vtb, 2);

  attn_kernel<<<(Bv * Hv * (Sv / 16)) / 8, 256, 0, stream>>>(qbuf, kbuf, vtb, ctx);

  gemm_nt_kernel<<<ggrid, 256, 0, stream>>>(ctx, wob, bo, (float*)d_out);
}
